// Network_4544075399258
// MI455X (gfx1250) — matrix operand order repaired and checked by reading, not compiled, not run on hardware
//
#include <hip/hip_runtime.h>
#include <math.h>

#define NN 50000
#define EE 800000
#define GG 512
#define HH 128
#define OUTC 10
#define LL 3
#define ETOT (EE + NN)
#define NHELEM ((size_t)NN * HH)
#define GHELEM ((size_t)GG * HH)
#define THRESH 0.01f

typedef __bf16 bf16;
typedef __attribute__((ext_vector_type(16))) __bf16 v16bf;
typedef __attribute__((ext_vector_type(8)))  __bf16 v8bf;
typedef __attribute__((ext_vector_type(8)))  float  v8f;

// global (addrspace 1) pointer types -> force global_load_* / global_store_*
typedef const v8bf  __attribute__((address_space(1)))* gv8bf;
typedef const v16bf __attribute__((address_space(1)))* gv16bf;
typedef float       __attribute__((address_space(1)))* gf32;

// ---------------- device helpers ----------------

__device__ __forceinline__ bf16 f2bf(float f) {
  unsigned u = __float_as_uint(f);
  unsigned r = (u + 0x7FFFu + ((u >> 16) & 1u)) >> 16;   // round-to-nearest-even
  unsigned short s = (unsigned short)r;
  bf16 b;
  __builtin_memcpy(&b, &s, 2);
  return b;
}

__device__ __forceinline__ float eluf(float x) {
  return x > 0.f ? x : (expf(x) - 1.f);
}

__device__ __forceinline__ void atomicMaxF(float* addr, float val) {
  int* ia = (int*)addr;
  int old = __float_as_int(*addr);
  while (__int_as_float(old) < val) {
    int assumed = old;
    old = atomicCAS(ia, assumed, __float_as_int(val));
    if (old == assumed) break;
  }
}

// ---------------- tiny kernels ----------------

__global__ void fill_f32(float* __restrict__ p, float v, size_t n) {
  size_t i = (size_t)blockIdx.x * blockDim.x + threadIdx.x;
  if (i < n) p[i] = v;
}

__global__ void zero_stats(double* stats) {
  stats[0] = 0.0;
  stats[1] = 0.0;
}

__device__ __forceinline__ void softK(const float* in, float* out, int K) {
  float m = -INFINITY;
  for (int k = 0; k < K; ++k) m = fmaxf(m, in[k]);
  float s = 0.f;
  for (int k = 0; k < K; ++k) { float e = expf(in[k] - m); out[k] = e; s += e; }
  float inv = 1.f / s;
  for (int k = 0; k < K; ++k) out[k] *= inv;
}

// alphas layout: na[12] | pool[9] | ro[12] | la[3]
__global__ void softmax_alphas(const float* __restrict__ na_log,
                               const float* __restrict__ pool_log,
                               const float* __restrict__ ro_log,
                               const float* __restrict__ la_log,
                               float* __restrict__ alphas) {
  int t = threadIdx.x;
  if (t < 3)       softK(na_log + t * 4,        alphas + t * 4,            4);
  else if (t < 6)  softK(pool_log + (t - 3) * 3, alphas + 12 + (t - 3) * 3, 3);
  else if (t < 10) softK(ro_log + (t - 6) * 3,   alphas + 21 + (t - 6) * 3, 3);
  else if (t == 10) softK(la_log,                alphas + 33,               3);
}

__global__ void count_nodes(const int* __restrict__ batch, float* __restrict__ cnt) {
  int i = blockIdx.x * blockDim.x + threadIdx.x;
  if (i < NN) atomicAdd(&cnt[batch[i]], 1.0f);
}

__global__ void cvt_bf16(const float* __restrict__ in, bf16* __restrict__ out, size_t n) {
  size_t i = (size_t)blockIdx.x * blockDim.x + threadIdx.x;
  if (i < n) out[i] = f2bf(in[i]);
}

// ---------------- weight fragment packing ----------------
// B-fragment layout for v_wmma_f32_16x16x32_bf16 (wave32):
// frag[((nt*4 + ki)*32 + lane)*16 + e] = W[K][Nc],
//   K = ki*32 + (lane>=16 ? 16 : 0) + e,  Nc = nt*16 + (lane & 15)

__global__ void prep_frag_single(const float* __restrict__ W, bf16* __restrict__ out) {
  int idx = blockIdx.x * blockDim.x + threadIdx.x;
  if (idx >= 16384) return;
  int e = idx & 15, lane = (idx >> 4) & 31, ki = (idx >> 9) & 3, nt = idx >> 11;
  int K = ki * 32 + (lane >> 4) * 16 + e;
  int Nc = nt * 16 + (lane & 15);
  out[idx] = f2bf(W[K * HH + Nc]);
}

__global__ void prep_frag_layer(const float* __restrict__ gW, const float* __restrict__ na,
                                bf16* __restrict__ w0f, bf16* __restrict__ whf,
                                bf16* __restrict__ wsnf, bf16* __restrict__ wsaf) {
  int idx = blockIdx.x * blockDim.x + threadIdx.x;
  if (idx >= 16384) return;
  int e = idx & 15, lane = (idx >> 4) & 31, ki = (idx >> 9) & 3, nt = idx >> 11;
  int K = ki * 32 + (lane >> 4) * 16 + e;
  int Nc = nt * 16 + (lane & 15);
  int p = K * HH + Nc;
  float a1 = na[1], a2 = na[2], a3 = na[3];
  w0f[idx]  = f2bf(gW[0 * 16384 + p]);
  whf[idx]  = f2bf(a1 * gW[2 * 16384 + p] + a2 * gW[3 * 16384 + p] + a3 * gW[4 * 16384 + p]);
  wsnf[idx] = f2bf(a1 * gW[1 * 16384 + p]);
  wsaf[idx] = f2bf(a2 * gW[3 * 16384 + p] + a3 * gW[5 * 16384 + p]);
}

// ---------------- WMMA GEMM: C[M x 128] = sum_t A_t[M x 128] @ B_t[128 x 128] ----------------
// 256 threads = 8 waves; block tile = 128 rows x 128 cols; wave w -> rows 16w..16w+15,
// all 8 col tiles held in registers (8 x v8f accumulators). NT terms fully unrolled,
// all fragment traffic via explicit global address space; B loads batched ahead of the
// 8-WMMA chain so waits are partial and the matrix pipe overlaps the next load clause.

template <int NT, bool EPI>
__global__ void __launch_bounds__(256)
gemm_wmma(const bf16* __restrict__ A0, const bf16* __restrict__ B0,
          const bf16* __restrict__ A1, const bf16* __restrict__ B1,
          const bf16* __restrict__ A2, const bf16* __restrict__ B2,
          const float* __restrict__ bias, float* __restrict__ C,
          int M, int act) {
  const int wave = threadIdx.x >> 5;
  const int lane = threadIdx.x & 31;
  const int rIn = lane & 15;
  const int half = lane >> 4;
  const int rowBase = blockIdx.x * 128 + wave * 16;

  int r = rowBase + rIn;
  int rA = r < M ? r : (M - 1);

  v8f acc[8];
#pragma unroll
  for (int i = 0; i < 8; ++i)
#pragma unroll
    for (int j = 0; j < 8; ++j) acc[i][j] = 0.f;

  const bf16* As[3] = {A0, A1, A2};
  const bf16* Bs[3] = {B0, B1, B2};

#pragma unroll
  for (int ki = 0; ki < 4; ++ki) {
#pragma unroll
    for (int t = 0; t < NT; ++t) {
      // A fragment: two contiguous 16B chunks per lane
      gv8bf ap = (gv8bf)(As[t] + (size_t)rA * HH + ki * 32 + half * 16);
      v8bf lo = ap[0];
      v8bf hi = ap[1];   // +8 bf16 = +16B
      v16bf a = __builtin_shufflevector(lo, hi, 0, 1, 2, 3, 4, 5, 6, 7,
                                        8, 9, 10, 11, 12, 13, 14, 15);
      // batch all 8 B-fragment loads, then run the WMMA chain
      const bf16* bbase = Bs[t] + ((size_t)ki * 32 + lane) * 16;
      v16bf bfr[8];
#pragma unroll
      for (int nt = 0; nt < 8; ++nt)
        bfr[nt] = *(gv16bf)(bbase + (size_t)nt * 2048);
#pragma unroll
      for (int nt = 0; nt < 8; ++nt)
        acc[nt] = __builtin_amdgcn_wmma_f32_16x16x32_bf16(
            false, a, false, bfr[nt], (short)0, acc[nt], false, false);
    }
  }

#pragma unroll
  for (int nt = 0; nt < 8; ++nt) {
    int col = nt * 16 + rIn;
#pragma unroll
    for (int v = 0; v < 8; ++v) {
      int rr = rowBase + v + half * 8;
      if (rr < M) {
        float val = acc[nt][v];
        if (EPI) {
          val += bias[col];
          if (act) val = eluf(val);
        }
        *(gf32)(C + (size_t)rr * HH + col) = val;
      }
    }
  }
}

// ---------------- sparse / edge kernels ----------------

__global__ void deg_kernel(const int* __restrict__ ei, const float* __restrict__ ew,
                           float* __restrict__ deg) {
  int e = blockIdx.x * blockDim.x + threadIdx.x;
  if (e >= ETOT) return;
  float w = ew[e];
  if (w == 0.f) return;
  int d = (e < EE) ? ei[EE + e] : (e - EE);
  atomicAdd(&deg[d], w);
}

__global__ void sagg_kernel(const int* __restrict__ ei, const float* __restrict__ ew,
                            const float* __restrict__ h, float* __restrict__ sagg) {
  int idx = blockIdx.x * blockDim.x + threadIdx.x;
  if (idx >= ETOT * 4) return;
  int e = idx >> 2, c = (idx & 3) * 32;
  float w = ew[e];
  if (w == 0.f) return;
  int s, d;
  if (e < EE) { s = ei[e]; d = ei[EE + e]; } else { s = e - EE; d = s; }
  const float4* hp = (const float4*)(h + (size_t)s * HH + c);
  float* op = sagg + (size_t)d * HH + c;
#pragma unroll
  for (int j = 0; j < 8; ++j) {
    float4 hv = hp[j];
    atomicAdd(&op[j * 4 + 0], hv.x * w);
    atomicAdd(&op[j * 4 + 1], hv.y * w);
    atomicAdd(&op[j * 4 + 2], hv.z * w);
    atomicAdd(&op[j * 4 + 3], hv.w * w);
  }
}

__global__ void gcn_kernel(const int* __restrict__ ei, const float* __restrict__ ew,
                           const float* __restrict__ deg, const float* __restrict__ hg,
                           const float* __restrict__ na, float* __restrict__ pre) {
  int idx = blockIdx.x * blockDim.x + threadIdx.x;
  if (idx >= ETOT * 4) return;
  int e = idx >> 2, c = (idx & 3) * 32;
  float w = ew[e];
  if (w == 0.f) return;
  int s, d;
  if (e < EE) { s = ei[e]; d = ei[EE + e]; } else { s = e - EE; d = s; }
  float sc = na[0] * w * rsqrtf(fmaxf(deg[s], 1e-6f)) * rsqrtf(fmaxf(deg[d], 1e-6f));
  const float4* hp = (const float4*)(hg + (size_t)s * HH + c);
  float* op = pre + (size_t)d * HH + c;
#pragma unroll
  for (int j = 0; j < 8; ++j) {
    float4 hv = hp[j];
    atomicAdd(&op[j * 4 + 0], hv.x * sc);
    atomicAdd(&op[j * 4 + 1], hv.y * sc);
    atomicAdd(&op[j * 4 + 2], hv.z * sc);
    atomicAdd(&op[j * 4 + 3], hv.w * sc);
  }
}

__global__ void prep_node(const float* __restrict__ sagg, const float* __restrict__ deg,
                          bf16* __restrict__ snb, bf16* __restrict__ sab) {
  size_t i = (size_t)blockIdx.x * blockDim.x + threadIdx.x;
  if (i >= NHELEM) return;
  int n = (int)(i >> 7);
  float v = sagg[i];
  sab[i] = f2bf(v);
  snb[i] = f2bf(v / fmaxf(deg[n], 1e-6f));
}

__global__ void update_ew(const float* __restrict__ keep, const int* __restrict__ ei,
                          float* __restrict__ ew) {
  int e = blockIdx.x * blockDim.x + threadIdx.x;
  if (e >= ETOT) return;
  float k;
  if (e < EE) k = keep[ei[e]] * keep[ei[EE + e]];
  else { float kk = keep[e - EE]; k = kk * kk; }
  ew[e] *= k;
}

// ---------------- elu + global layernorm stats ----------------

__global__ void elu_stats(const float* __restrict__ pre, float* __restrict__ h,
                          double* __restrict__ stats) {
  size_t i = (size_t)blockIdx.x * blockDim.x + threadIdx.x;
  size_t stride = (size_t)gridDim.x * blockDim.x;
  double s = 0.0, sq = 0.0;
  for (; i < NHELEM; i += stride) {
    float e = eluf(pre[i]);
    h[i] = e;
    s += (double)e;
    sq += (double)e * (double)e;
  }
  for (int off = 16; off; off >>= 1) {
    s  += __shfl_xor(s, off);
    sq += __shfl_xor(sq, off);
  }
  if ((threadIdx.x & 31) == 0) {
    atomicAdd(&stats[0], s);
    atomicAdd(&stats[1], sq);
  }
}

__global__ void finalize_stats(const double* __restrict__ stats, float* __restrict__ mi) {
  double cnt = (double)NHELEM;
  double mean = stats[0] / cnt;
  double var = stats[1] / cnt - mean * mean;
  mi[0] = (float)mean;
  mi[1] = (float)(1.0 / sqrt(var + 1e-5));
}

// ---------------- pooling gates (one wave per node) ----------------

__global__ void pool_gate(const float* __restrict__ mi, const float* __restrict__ p0,
                          const float* __restrict__ p1, const float* __restrict__ pa,
                          float* __restrict__ h, bf16* __restrict__ hb,
                          float* __restrict__ keep) {
  int wave = threadIdx.x >> 5, lane = threadIdx.x & 31;
  int n = blockIdx.x * 8 + wave;
  if (n >= NN) return;
  float mean = mi[0], inv = mi[1];
  float* hrow = h + (size_t)n * HH + lane * 4;
  float4 hv = *(const float4*)hrow;
  float hn0 = (hv.x - mean) * inv, hn1 = (hv.y - mean) * inv;
  float hn2 = (hv.z - mean) * inv, hn3 = (hv.w - mean) * inv;
  float4 q0 = *(const float4*)(p0 + lane * 4);
  float4 q1 = *(const float4*)(p1 + lane * 4);
  float d1 = hn0 * q0.x + hn1 * q0.y + hn2 * q0.z + hn3 * q0.w;
  float d2 = hn0 * q1.x + hn1 * q1.y + hn2 * q1.z + hn3 * q1.w;
  for (int off = 16; off; off >>= 1) {
    d1 += __shfl_xor(d1, off);
    d2 += __shfl_xor(d2, off);
  }
  float gate = pa[0] * (1.f / (1.f + expf(-d1))) + pa[1] * tanhf(d2) + pa[2];
  float k = gate > THRESH ? 1.f : 0.f;
  float sc = gate * k;
  hn0 *= sc; hn1 *= sc; hn2 *= sc; hn3 *= sc;
  float4 o; o.x = hn0; o.y = hn1; o.z = hn2; o.w = hn3;
  *(float4*)hrow = o;
  bf16* bp = hb + (size_t)n * HH + lane * 4;
  bp[0] = f2bf(hn0); bp[1] = f2bf(hn1); bp[2] = f2bf(hn2); bp[3] = f2bf(hn3);
  if (lane == 0) keep[n] = k;
}

// ---------------- readouts ----------------

__global__ void readout_scatter(const float* __restrict__ h, const int* __restrict__ batch,
                                float* __restrict__ ssum, float* __restrict__ smax) {
  size_t i = (size_t)blockIdx.x * blockDim.x + threadIdx.x;
  if (i >= NHELEM) return;
  int n = (int)(i >> 7), f = (int)(i & 127);
  int g = batch[n];
  float v = h[i];
  atomicAdd(&ssum[g * HH + f], v);
  atomicMaxF(&smax[g * HH + f], v);
}

__global__ void readout_combine(const float* __restrict__ ssum, const float* __restrict__ smax,
                                const float* __restrict__ cnt, const float* __restrict__ w3,
                                float* __restrict__ rep) {
  int i = blockIdx.x * blockDim.x + threadIdx.x;
  if (i >= (int)GHELEM) return;
  int g = i >> 7;
  float c = fmaxf(cnt[g], 1.f);
  float mx = smax[i];
  if (!isfinite(mx)) mx = 0.f;
  float s = ssum[i];
  rep[i] = w3[0] * (s / c) + w3[1] * mx + w3[2] * s;
}

// ---------------- output head ----------------

__global__ void final_combine(const float* __restrict__ reps, const float* __restrict__ la,
                              float* __restrict__ z) {
  int i = blockIdx.x * blockDim.x + threadIdx.x;
  if (i >= (int)GHELEM) return;
  float s = 0.f, mx = -INFINITY;
  for (int l = 0; l < LL + 1; ++l) {
    float v = reps[(size_t)l * GHELEM + i];
    s += v;
    mx = fmaxf(mx, v);
  }
  z[i] = la[0] * eluf(s) + la[1] * eluf(s * (1.f / (LL + 1))) + la[2] * eluf(mx);
}

__global__ void lin_out_kernel(const float* __restrict__ z, const float* __restrict__ W,
                               const float* __restrict__ b, float* __restrict__ z2) {
  int i = blockIdx.x * blockDim.x + threadIdx.x;
  if (i >= (int)GHELEM) return;
  int g = i >> 7, oh = i & 127;
  float acc = b[oh];
  const float* zr = z + (size_t)g * HH;
  for (int k = 0; k < HH; ++k) acc += zr[k] * W[k * HH + oh];
  z2[i] = eluf(acc);
}

__global__ void cls_kernel(const float* __restrict__ z2, const float* __restrict__ W,
                           const float* __restrict__ b, float* __restrict__ out) {
  int wave = threadIdx.x >> 5, lane = threadIdx.x & 31;
  int g = blockIdx.x * 8 + wave;
  if (g >= GG) return;
  float lg = -INFINITY;
  if (lane < OUTC) {
    float acc = b[lane];
    const float* zr = z2 + (size_t)g * HH;
    for (int k = 0; k < HH; ++k) acc += zr[k] * W[k * OUTC + lane];
    lg = acc;
  }
  float m = lg;
  for (int off = 16; off; off >>= 1) m = fmaxf(m, __shfl_xor(m, off));
  float ex = (lane < OUTC) ? expf(lg - m) : 0.f;
  float se = ex;
  for (int off = 16; off; off >>= 1) se += __shfl_xor(se, off);
  if (lane < OUTC) out[g * OUTC + lane] = lg - m - logf(se);
}

// ---------------- host ----------------

static inline int cdiv(long long a, long long b) { return (int)((a + b - 1) / b); }

extern "C" void kernel_launch(void* const* d_in, const int* in_sizes, int n_in,
                              void* d_out, int out_size, void* d_ws, size_t ws_size,
                              hipStream_t stream) {
  (void)in_sizes; (void)n_in; (void)out_size; (void)ws_size;

  const float* x       = (const float*)d_in[0];
  const int*   ei      = (const int*)d_in[1];
  const int*   batch   = (const int*)d_in[2];
  const float* lin1_W  = (const float*)d_in[3];
  const float* lin1_b  = (const float*)d_in[4];
  const float* gnn_W   = (const float*)d_in[5];
  const float* pool_p  = (const float*)d_in[6];
  const float* lout_W  = (const float*)d_in[7];
  const float* lout_b  = (const float*)d_in[8];
  const float* cls_W   = (const float*)d_in[9];
  const float* cls_b   = (const float*)d_in[10];
  const float* na_log  = (const float*)d_in[11];
  const float* pool_log= (const float*)d_in[12];
  const float* ro_log  = (const float*)d_in[13];
  const float* la_log  = (const float*)d_in[14];
  float* out = (float*)d_out;

  char* w = (char*)d_ws;
  size_t o = 0;
  auto take = [&](size_t bytes) { size_t r = o; o += (bytes + 255) & ~(size_t)255; return r; };

  float*  h    = (float*)(w + take(NHELEM * 4));
  float*  pre  = (float*)(w + take(NHELEM * 4));
  float*  hg   = (float*)(w + take(NHELEM * 4));
  float*  sagg = (float*)(w + take(NHELEM * 4));
  bf16*   hb   = (bf16*)(w + take(NHELEM * 2));
  bf16*   snb  = (bf16*)(w + take(NHELEM * 2));
  bf16*   sab  = (bf16*)(w + take(NHELEM * 2));
  float*  deg  = (float*)(w + take((size_t)NN * 4));
  float*  keep = (float*)(w + take((size_t)NN * 4));
  float*  ew   = (float*)(w + take((size_t)ETOT * 4));
  bf16*   w0f  = (bf16*)(w + take(16384 * 2));
  bf16*   whf  = (bf16*)(w + take(16384 * 2));
  bf16*   wsnf = (bf16*)(w + take(16384 * 2));
  bf16*   wsaf = (bf16*)(w + take(16384 * 2));
  float*  alphas = (float*)(w + take(64 * 4));
  double* stats  = (double*)(w + take(4 * 8));
  float*  mi     = (float*)(w + take(4 * 4));
  float*  reps = (float*)(w + take((size_t)(LL + 1) * GHELEM * 4));
  float*  ssum = (float*)(w + take(GHELEM * 4));
  float*  smax = (float*)(w + take(GHELEM * 4));
  float*  cnt  = (float*)(w + take((size_t)GG * 4));
  float*  z    = (float*)(w + take(GHELEM * 4));
  float*  z2   = (float*)(w + take(GHELEM * 4));

  const int T = 256;

  // alphas, edge weights, per-graph node counts
  softmax_alphas<<<1, 32, 0, stream>>>(na_log, pool_log, ro_log, la_log, alphas);
  fill_f32<<<cdiv(ETOT, T), T, 0, stream>>>(ew, 1.f, (size_t)ETOT);
  fill_f32<<<cdiv(GG, T), T, 0, stream>>>(cnt, 0.f, (size_t)GG);
  count_nodes<<<cdiv(NN, T), T, 0, stream>>>(batch, cnt);

  // h = elu(x @ lin1_W + b), via WMMA
  cvt_bf16<<<cdiv((long long)NHELEM, T), T, 0, stream>>>(x, hb, NHELEM);
  prep_frag_single<<<64, T, 0, stream>>>(lin1_W, w0f);
  gemm_wmma<1, true><<<cdiv(NN, 128), T, 0, stream>>>(
      hb, w0f, nullptr, nullptr, nullptr, nullptr, lin1_b, h, NN, 1);
  cvt_bf16<<<cdiv((long long)NHELEM, T), T, 0, stream>>>(h, hb, NHELEM);

  // readout 0
  fill_f32<<<cdiv((long long)GHELEM, T), T, 0, stream>>>(ssum, 0.f, GHELEM);
  fill_f32<<<cdiv((long long)GHELEM, T), T, 0, stream>>>(smax, -INFINITY, GHELEM);
  readout_scatter<<<cdiv((long long)NHELEM, T), T, 0, stream>>>(h, batch, ssum, smax);
  readout_combine<<<cdiv((long long)GHELEM, T), T, 0, stream>>>(ssum, smax, cnt,
                                                               alphas + 21, reps);

  for (int l = 0; l < LL; ++l) {
    prep_frag_layer<<<64, T, 0, stream>>>(gnn_W + (size_t)l * 6 * 16384, alphas + l * 4,
                                          w0f, whf, wsnf, wsaf);
    fill_f32<<<cdiv(NN, T), T, 0, stream>>>(deg, 0.f, (size_t)NN);
    fill_f32<<<cdiv((long long)NHELEM, T), T, 0, stream>>>(sagg, 0.f, NHELEM);
    deg_kernel<<<cdiv(ETOT, T), T, 0, stream>>>(ei, ew, deg);
    sagg_kernel<<<cdiv((long long)ETOT * 4, T), T, 0, stream>>>(ei, ew, h, sagg);
    prep_node<<<cdiv((long long)NHELEM, T), T, 0, stream>>>(sagg, deg, snb, sab);

    // hg = h @ W0 ; pre = h@Wh + (sagg/deg)@Wsn + sagg@Wsa
    gemm_wmma<1, false><<<cdiv(NN, 128), T, 0, stream>>>(
        hb, w0f, nullptr, nullptr, nullptr, nullptr, nullptr, hg, NN, 0);
    gemm_wmma<3, false><<<cdiv(NN, 128), T, 0, stream>>>(
        hb, whf, snb, wsnf, sab, wsaf, nullptr, pre, NN, 0);
    // pre += a0 * sym-normalized scatter of hg
    gcn_kernel<<<cdiv((long long)ETOT * 4, T), T, 0, stream>>>(ei, ew, deg, hg,
                                                               alphas + l * 4, pre);
    // elu + full-tensor layernorm stats
    zero_stats<<<1, 1, 0, stream>>>(stats);
    elu_stats<<<2048, T, 0, stream>>>(pre, h, stats);
    finalize_stats<<<1, 1, 0, stream>>>(stats, mi);
    // normalize + gates + mask, write fp32 + bf16 h
    pool_gate<<<cdiv(NN, 8), T, 0, stream>>>(mi, pool_p + (size_t)(l * 2) * HH,
                                             pool_p + (size_t)(l * 2 + 1) * HH,
                                             alphas + 12 + l * 3, h, hb, keep);
    update_ew<<<cdiv(ETOT, T), T, 0, stream>>>(keep, ei, ew);

    fill_f32<<<cdiv((long long)GHELEM, T), T, 0, stream>>>(ssum, 0.f, GHELEM);
    fill_f32<<<cdiv((long long)GHELEM, T), T, 0, stream>>>(smax, -INFINITY, GHELEM);
    readout_scatter<<<cdiv((long long)NHELEM, T), T, 0, stream>>>(h, batch, ssum, smax);
    readout_combine<<<cdiv((long long)GHELEM, T), T, 0, stream>>>(
        ssum, smax, cnt, alphas + 21 + (l + 1) * 3, reps + (size_t)(l + 1) * GHELEM);
  }

  final_combine<<<cdiv((long long)GHELEM, T), T, 0, stream>>>(reps, alphas + 33, z);
  lin_out_kernel<<<cdiv((long long)GHELEM, T), T, 0, stream>>>(z, lout_W, lout_b, z2);
  cls_kernel<<<cdiv(GG, 8), T, 0, stream>>>(z2, cls_W, cls_b, out);
}
